// RNN_MoO_time_31516470018754
// MI455X (gfx1250) — compile-verified, matrix operand order checked by reading, not run on hardware
//
#include <hip/hip_runtime.h>
#include <hip/hip_bf16.h>

// ---------------------------------------------------------------------------
// Mixture-of-Outputs LSTM on gfx1250 (MI455X), f16 WMMA path, fused per-step.
// SEQ=256, BS=64, IN=512, HID=1024, K=4 experts, gates i,f,g,o (4H=4096).
// Per step t:  gates[k] = [x_t | h] (64x1536) @ Wcat[k]^T + (bih+bhh)[k]
//              c_k = sig(f)*c + sig(i)*tanh(g);  h_k = sig(o)*tanh(c_k)
//              h,c = sum_k softmax(coef[t])_k * {h_k, c_k};  outs[t] = h
// One fused kernel per step: GEMM (WMMA) + pointwise + expert mixture.
// ---------------------------------------------------------------------------

#define SEQ 256
#define BS  64
#define IN  512
#define HID 1024
#define KEXP 4
#define KDIM (IN + HID)          // 1536
#define GH  (4 * HID)            // 4096 rows per expert
#define WROWS (KEXP * GH)        // 16384 rows total
#define BH  (BS * HID)           // 65536 elements of h/c state

typedef _Float16 f16_t;
typedef __attribute__((ext_vector_type(16))) _Float16 v16h;
typedef __attribute__((ext_vector_type(8)))  _Float16 v8h;    // 16 bytes
typedef __attribute__((ext_vector_type(8)))  float    v8f;

// Load a 16x32 (16-bit) WMMA A/B fragment from a row-major tile.
// Per the CDNA5 ISA 16-bit operand layout:
//   lanes 0-15 : row=lane,    halves 0-7 = K+0..7,  halves 8-15 = K+16..23
//   lanes 16-31: row=lane-16, halves 0-7 = K+8..15, halves 8-15 = K+24..31
__device__ __forceinline__ v16h load_frag_rm(const f16_t* base, int ld) {
  const int lane = __builtin_amdgcn_workitem_id_x() & 31;
  const int row  = lane & 15;
  const int koff = (lane >> 4) << 3;   // 0 or 8
  const f16_t* p = base + row * ld + koff;
  union { v8h h[2]; v16h w; } u;
  u.h[0] = *(const v8h*)(p);           // K+koff    .. K+koff+7
  u.h[1] = *(const v8h*)(p + 16);      // K+koff+16 .. K+koff+23
  return u.w;
}

__device__ __forceinline__ v8f wmma_f16(v16h a, v16h b, v8f c) {
  return __builtin_amdgcn_wmma_f32_16x16x32_f16(
      false, a, false, b, (short)0, c, false, false);
}

__device__ __forceinline__ float sigmoidf_fast(float v) {
  return 1.0f / (1.0f + __expf(-v));
}

// ---------------------------------------------------------------------------
// Prep 1: pack [Wih | Whh] fp32 -> f16, row-major [16384 x 1536].
// Row = (k*4 + gate)*1024 + h (PyTorch gate order i,f,g,o).
// ---------------------------------------------------------------------------
__global__ __launch_bounds__(256) void pack_w_kernel(
    const float* __restrict__ Wih, const float* __restrict__ Whh,
    f16_t* __restrict__ wcat) {
  size_t i = (size_t)blockIdx.x * 256 + threadIdx.x;   // 0 .. 16384*1536-1
  int col = (int)(i % KDIM);
  int row = (int)(i / KDIM);
  int k   = row >> 12;          // /4096
  int r4  = row & 4095;         // gate*1024 + h
  float v = (col < IN) ? Wih[((size_t)k * GH + r4) * IN + col]
                       : Whh[((size_t)k * GH + r4) * HID + (col - IN)];
  wcat[i] = (f16_t)v;
}

// ---------------------------------------------------------------------------
// Prep 2: x -> f16, bias = bih + bhh, init h buffer 0 (f16) and c state (f32).
// ---------------------------------------------------------------------------
__global__ __launch_bounds__(256) void prep_misc_kernel(
    const float* __restrict__ x,   const float* __restrict__ bih,
    const float* __restrict__ bhh, const float* __restrict__ h0,
    const float* __restrict__ c0,
    f16_t* __restrict__ xf16, float* __restrict__ bias,
    f16_t* __restrict__ hbuf0, float* __restrict__ cst) {
  size_t i = (size_t)blockIdx.x * 256 + threadIdx.x;
  const size_t nx = (size_t)SEQ * BS * IN;             // 8388608
  if (i < nx)            xf16[i] = (f16_t)x[i];
  if (i < (size_t)WROWS) bias[i] = bih[i] + bhh[i];
  if (i < (size_t)BH) { hbuf0[i] = (f16_t)h0[i]; cst[i] = c0[i]; }
}

// ---------------------------------------------------------------------------
// Fused step kernel. Grid: 128 blocks = 64 h-slices x 2 batch-halves.
// Block = 256 threads (8 waves). Each block handles ALL 4 experts for its
// (32-batch-row, 16-h-col) tile, so the LSTM pointwise AND the softmax
// mixture are done locally; only f16 h (ping-pong) crosses blocks.
//
// Wave w (0..7): lmt = w>>2 (local 16-row M tile), kp = (w>>1)&1 (expert
// pair), gp = w&1 (gate pair). 4 accumulators share one A fragment per
// K-chunk: 4 WMMAs per (1 A + 4 B) fragment loads.
// K-loop is split x-phase / h-phase (no per-iteration operand select).
// ---------------------------------------------------------------------------
__global__ __launch_bounds__(256) void step_fused_kernel(
    const f16_t* __restrict__ xf16,   // [SEQ][64][512]
    const f16_t* __restrict__ hsrc,   // [64][1024]  f16 h state (read)
    f16_t* __restrict__ hdst,         // [64][1024]  f16 h state (write)
    const f16_t* __restrict__ wcat,   // [16384][1536]
    const float* __restrict__ bias,   // [16384] = [k][gate*1024 + h]
    float* __restrict__ cst,          // [64][1024] f32 c state (slice-owned)
    const float* __restrict__ coef,   // [SEQ][4]
    float* __restrict__ out,          // [SEQ][64][1024] ++ h[64][1024] ++ c[...]
    int t) {
  // [gate][expert][b_local 0..31][n 0..15]; reused in-place for h_k / c_k.
  __shared__ float lds_g[4][KEXP][32][16];   // 32 KB

  const int hb    = blockIdx.x & 63;        // h-slice: cols hb*16 .. +16
  const int bhalf = blockIdx.x >> 6;        // batch half: rows bhalf*32 .. +32
  const int wave  = threadIdx.x >> 5;
  const int lmt   = wave >> 2;              // local M tile (0..1)
  const int ke0   = ((wave >> 1) & 1) * 2;  // experts ke0, ke0+1
  const int g0    = (wave & 1) * 2;         // gates  g0,  g0+1
  const int mrow  = bhalf * 32 + lmt * 16;  // global batch row of this M tile

  const f16_t* xt = xf16 + ((size_t)t * BS + mrow) * IN;
  const f16_t* hp = hsrc + (size_t)mrow * HID;
  const f16_t* w00 = wcat + ((size_t)((ke0)*4 + g0    ) * HID + hb * 16) * KDIM;
  const f16_t* w01 = wcat + ((size_t)((ke0)*4 + g0 + 1) * HID + hb * 16) * KDIM;
  const f16_t* w10 = wcat + ((size_t)((ke0+1)*4 + g0    ) * HID + hb * 16) * KDIM;
  const f16_t* w11 = wcat + ((size_t)((ke0+1)*4 + g0 + 1) * HID + hb * 16) * KDIM;

  v8f a00 = {}, a01 = {}, a10 = {}, a11 = {};

  // ---- x phase: K = 0..512 ----
#pragma unroll 4
  for (int k0 = 0; k0 < IN; k0 += 32) {
    v16h a  = load_frag_rm(xt  + k0, IN);
    v16h b0 = load_frag_rm(w00 + k0, KDIM);
    v16h b1 = load_frag_rm(w01 + k0, KDIM);
    v16h b2 = load_frag_rm(w10 + k0, KDIM);
    v16h b3 = load_frag_rm(w11 + k0, KDIM);
    a00 = wmma_f16(a, b0, a00);
    a01 = wmma_f16(a, b1, a01);
    a10 = wmma_f16(a, b2, a10);
    a11 = wmma_f16(a, b3, a11);
  }
  // ---- h phase: K = 512..1536 ----
#pragma unroll 4
  for (int k0 = 0; k0 < HID; k0 += 32) {
    v16h a  = load_frag_rm(hp  + k0,      HID);
    v16h b0 = load_frag_rm(w00 + IN + k0, KDIM);
    v16h b1 = load_frag_rm(w01 + IN + k0, KDIM);
    v16h b2 = load_frag_rm(w10 + IN + k0, KDIM);
    v16h b3 = load_frag_rm(w11 + IN + k0, KDIM);
    a00 = wmma_f16(a, b0, a00);
    a01 = wmma_f16(a, b1, a01);
    a10 = wmma_f16(a, b2, a10);
    a11 = wmma_f16(a, b3, a11);
  }

  // Scatter D tiles to LDS. f32 C/D layout: lane&15 = column N;
  // vgpr r = row r (lanes 0-15) or row r+8 (lanes 16-31).
  {
    const int lane  = threadIdx.x & 31;
    const int n     = lane & 15;
    const int rbase = (lane >> 4) * 8;
#pragma unroll
    for (int r = 0; r < 8; ++r) {
      const int b = lmt * 16 + rbase + r;
      lds_g[g0    ][ke0    ][b][n] = a00[r];
      lds_g[g0 + 1][ke0    ][b][n] = a01[r];
      lds_g[g0    ][ke0 + 1][b][n] = a10[r];
      lds_g[g0 + 1][ke0 + 1][b][n] = a11[r];
    }
  }
  __syncthreads();

  // Pointwise LSTM cell per expert element: 4*32*16 = 2048 elems.
  // In-place: h_k -> lds_g[0][ke][b][n], c_k -> lds_g[1][ke][b][n]
  // (each element's 4 gate reads happen before its 2 writes; disjoint elems).
  for (int e = threadIdx.x; e < KEXP * 32 * 16; e += 256) {
    const int ke = e >> 9;
    const int b  = (e >> 4) & 31;
    const int n  = e & 15;
    const int hcol = hb * 16 + n;
    float gi = lds_g[0][ke][b][n] + bias[(ke * 4 + 0) * HID + hcol];
    float gf = lds_g[1][ke][b][n] + bias[(ke * 4 + 1) * HID + hcol];
    float gg = lds_g[2][ke][b][n] + bias[(ke * 4 + 2) * HID + hcol];
    float go = lds_g[3][ke][b][n] + bias[(ke * 4 + 3) * HID + hcol];
    gi = sigmoidf_fast(gi);
    gf = sigmoidf_fast(gf);
    gg = tanhf(gg);
    go = sigmoidf_fast(go);
    const float cn = gf * cst[(bhalf * 32 + b) * HID + hcol] + gi * gg;
    const float hn = go * tanhf(cn);
    lds_g[0][ke][b][n] = hn;   // h_k
    lds_g[1][ke][b][n] = cn;   // c_k
  }
  __syncthreads();

  // Softmax(coef[t]) mixture over experts: 32*16 = 512 (b,n) pairs.
  const float c0v = coef[t * 4 + 0], c1v = coef[t * 4 + 1];
  const float c2v = coef[t * 4 + 2], c3v = coef[t * 4 + 3];
  const float mx  = fmaxf(fmaxf(c0v, c1v), fmaxf(c2v, c3v));
  const float e0 = __expf(c0v - mx), e1 = __expf(c1v - mx);
  const float e2 = __expf(c2v - mx), e3 = __expf(c3v - mx);
  const float inv = 1.0f / (e0 + e1 + e2 + e3);
  const float w0 = e0 * inv, w1 = e1 * inv, w2 = e2 * inv, w3 = e3 * inv;

  for (int e = threadIdx.x; e < 32 * 16; e += 256) {
    const int b = e >> 4;
    const int n = e & 15;
    const float hn = w0 * lds_g[0][0][b][n] + w1 * lds_g[0][1][b][n] +
                     w2 * lds_g[0][2][b][n] + w3 * lds_g[0][3][b][n];
    const float cn = w0 * lds_g[1][0][b][n] + w1 * lds_g[1][1][b][n] +
                     w2 * lds_g[1][2][b][n] + w3 * lds_g[1][3][b][n];
    const size_t o = (size_t)(bhalf * 32 + b) * HID + hb * 16 + n;
    out[(size_t)t * BH + o] = hn;       // outs[t]
    hdst[o] = (f16_t)hn;                // GEMM operand for step t+1
    cst[o]  = cn;
    if (t == SEQ - 1) {
      out[(size_t)SEQ * BH + o]       = hn;   // final h [1,B,H]
      out[(size_t)(SEQ + 1) * BH + o] = cn;   // final c [1,B,H]
    }
  }
}

// ---------------------------------------------------------------------------
// Workspace layout (bytes):
//   wcat  f16 [16384*1536]  @ 0         (50331648)
//   xf16  f16 [256*64*512]  @ 50331648  (16777216)
//   bias  f32 [16384]       @ 67108864  (65536)
//   hbuf0 f16 [64*1024]     @ 67174400  (131072)
//   hbuf1 f16 [64*1024]     @ 67305472  (131072)
//   cst   f32 [64*1024]     @ 67436544  (262144)
// total ~67.7 MB
// ---------------------------------------------------------------------------
extern "C" void kernel_launch(void* const* d_in, const int* in_sizes, int n_in,
                              void* d_out, int out_size, void* d_ws, size_t ws_size,
                              hipStream_t stream) {
  (void)in_sizes; (void)n_in; (void)out_size; (void)ws_size;
  const float* x    = (const float*)d_in[0];
  const float* h0   = (const float*)d_in[1];
  const float* c0   = (const float*)d_in[2];
  const float* coef = (const float*)d_in[3];
  const float* Wih  = (const float*)d_in[4];
  const float* Whh  = (const float*)d_in[5];
  const float* bih  = (const float*)d_in[6];
  const float* bhh  = (const float*)d_in[7];

  char* ws = (char*)d_ws;
  f16_t* wcat  = (f16_t*)(ws + 0);
  f16_t* xf16  = (f16_t*)(ws + 50331648);
  float* bias  = (float*)(ws + 67108864);
  f16_t* hbuf0 = (f16_t*)(ws + 67174400);
  f16_t* hbuf1 = (f16_t*)(ws + 67305472);
  float* cst   = (float*)(ws + 67436544);
  float* out   = (float*)d_out;

  pack_w_kernel<<<(WROWS * (size_t)KDIM) / 256, 256, 0, stream>>>(Wih, Whh, wcat);
  prep_misc_kernel<<<((size_t)SEQ * BS * IN) / 256, 256, 0, stream>>>(
      x, bih, bhh, h0, c0, xf16, bias, hbuf0, cst);

  for (int t = 0; t < SEQ; ++t) {
    const f16_t* hsrc = (t & 1) ? hbuf1 : hbuf0;
    f16_t*       hdst = (t & 1) ? hbuf0 : hbuf1;
    step_fused_kernel<<<128, 256, 0, stream>>>(xf16, hsrc, hdst, wcat, bias,
                                               cst, coef, out, t);
  }
}
